// FlowHAB_90709709291797
// MI455X (gfx1250) — matrix operand order repaired and checked by reading, not compiled, not run on hardware
//
#include <hip/hip_runtime.h>

#define Bsz 8
#define Cch 192
#define NHEADS 6
#define DHd 32
#define SHF 4
#define TDIM 512
#define NGRP 32
#define CPG 6       // channels per group
#define HIDd 384
#define GATEH 24
#define HWp 16384   // 128*128

typedef __bf16 bhalf;
typedef __attribute__((ext_vector_type(16))) __bf16 v16bf;
typedef __attribute__((ext_vector_type(8)))  float  v8f;

__device__ __forceinline__ v8f wmma_bf16(v16bf a, v16bf b, v8f c) {
  return __builtin_amdgcn_wmma_f32_16x16x32_bf16(false, a, false, b, (short)0, c,
                                                 false, false);
}

// ---- fragment loaders (contiguous k = 16*(lane/16) + s order, wave32) ----
// A: 16x32 (MxK) 16-bit. lane 0-15: M=lane, K=0..15; lanes 16-31: K=16..31.
__device__ __forceinline__ v16bf ldA_row(const bhalf* __restrict__ r) {
  v16bf a;
#pragma unroll
  for (int v = 0; v < 8; ++v) {
    const int kk = 2 * v;
    a[2 * v]     = r[kk];
    a[2 * v + 1] = r[kk + 1];
  }
  return a;
}
__device__ __forceinline__ v16bf ldA(const bhalf* __restrict__ s, int ld, int m0,
                                     int k0, int lane) {
  return ldA_row(s + (m0 + (lane & 15)) * ld + k0 + ((lane >> 4) << 4));
}
// B: 32x16 (KxN). lanes 0-15 hold K=0..15, lanes 16-31 hold K=16..31; N = lane%16.
// ldBt: source stored row-major [N][K] (weight W[n][k]) -> contiguous K per lane.
__device__ __forceinline__ v16bf ldBt(const bhalf* __restrict__ s, int ld, int n0,
                                      int k0, int lane) {
  const bhalf* r = s + (n0 + (lane & 15)) * ld + k0 + ((lane >> 4) << 4);
  v16bf b;
#pragma unroll
  for (int i = 0; i < 16; ++i) b[i] = r[i];
  return b;
}
// ldBn: source stored row-major [K][N].
__device__ __forceinline__ v16bf ldBn(const bhalf* __restrict__ s, int ld, int n0,
                                      int k0, int lane) {
  const int n  = n0 + (lane & 15);
  const int kb = k0 + ((lane >> 4) << 4);
  v16bf b;
#pragma unroll
  for (int i = 0; i < 16; ++i) b[i] = s[(kb + i) * ld + n];
  return b;
}

// ---------------- prep: fp32 -> bf16 weights, conv repack, zero pooled ----------
__global__ void k_prep(const float* __restrict__ ain_w, const float* __restrict__ aout_w,
                       const float* __restrict__ proj_w, const float* __restrict__ f1_w,
                       const float* __restrict__ f2_w, const float* __restrict__ cc_w,
                       bhalf* w_in, bhalf* w_out, bhalf* w_proj, bhalf* w_f1,
                       bhalf* w_f2, bhalf* w_cc, float* pooled) {
  const int i = blockIdx.x * blockDim.x + threadIdx.x;
  const int stride = gridDim.x * blockDim.x;
  for (int t = i; t < 576 * 192; t += stride) w_in[t] = (bhalf)ain_w[t];
  for (int t = i; t < 192 * 192; t += stride) w_out[t] = (bhalf)aout_w[t];
  for (int t = i; t < 192 * 192; t += stride) w_proj[t] = (bhalf)proj_w[t];
  for (int t = i; t < 384 * 192; t += stride) w_f1[t] = (bhalf)f1_w[t];
  for (int t = i; t < 192 * 384; t += stride) w_f2[t] = (bhalf)f2_w[t];
  for (int t = i; t < 9 * 192 * 192; t += stride) {
    const int tap = t / (192 * 192), rem = t % (192 * 192);
    const int o = rem / 192, c = rem % 192;
    const int ky = tap / 3, kx = tap % 3;
    w_cc[t] = (bhalf)cc_w[((o * 192 + c) * 3 + ky) * 3 + kx];
  }
  for (int t = i; t < Bsz * 192; t += stride) pooled[t] = 0.f;
}

// ---------------- time-affine params: p = t_emb @ ta_w.T + ta_b -----------------
__global__ void k_taffine(const float* __restrict__ t_emb,
                          const float* __restrict__ ta1_w, const float* __restrict__ ta1_b,
                          const float* __restrict__ ta2_w, const float* __restrict__ ta2_b,
                          float* p1, float* p2) {
  const int i = blockIdx.x * blockDim.x + threadIdx.x;
  if (i >= 2 * Bsz * 384) return;
  const int which = i / (Bsz * 384);
  const int r = i % (Bsz * 384);
  const int b = r / 384, d = r % 384;
  const float* w = which ? ta2_w : ta1_w;
  const float* bb = which ? ta2_b : ta1_b;
  const float* te = t_emb + b * TDIM;
  float acc = bb[d];
  for (int k = 0; k < TDIM; ++k) acc += te[k] * w[d * TDIM + k];
  (which ? p2 : p1)[b * 384 + d] = acc;
}

// ---------------- groupnorm stats --------------------------------------------
__global__ void k_gnstats(const float* __restrict__ src, float* mean, float* rstd) {
  const int bg = blockIdx.x;  // b*32 + g
  const int b = bg / NGRP, g = bg % NGRP;
  const float* base = src + ((size_t)b * Cch + g * CPG) * HWp;
  float s = 0.f, s2 = 0.f;
  for (int t = threadIdx.x; t < CPG * HWp; t += blockDim.x) {
    const float v = base[t];
    s += v; s2 += v * v;
  }
  __shared__ float sh0[256], sh1[256];
  sh0[threadIdx.x] = s; sh1[threadIdx.x] = s2;
  __syncthreads();
  for (int o = 128; o > 0; o >>= 1) {
    if (threadIdx.x < o) {
      sh0[threadIdx.x] += sh0[threadIdx.x + o];
      sh1[threadIdx.x] += sh1[threadIdx.x + o];
    }
    __syncthreads();
  }
  if (threadIdx.x == 0) {
    const float inv = 1.f / (float)(CPG * HWp);
    const float m = sh0[0] * inv;
    const float var = sh1[0] * inv - m * m;
    mean[bg] = m;
    rstd[bg] = rsqrtf(var + 1e-5f);
  }
}

// ---------------- fused window attention + proj + residual -------------------
__global__ __launch_bounds__(256) void k_attn(
    const float* __restrict__ xin, const float* __restrict__ p1,
    const float* __restrict__ n1w, const float* __restrict__ n1b,
    const float* __restrict__ mean1, const float* __restrict__ rstd1,
    const bhalf* __restrict__ w_in, const float* __restrict__ b_in,
    const bhalf* __restrict__ w_out, const float* __restrict__ b_out,
    const bhalf* __restrict__ w_proj, const float* __restrict__ b_proj,
    float* __restrict__ x1) {
  __shared__ bhalf Xw[64 * 192];    // window input; reused as attention-O buffer
  __shared__ bhalf QKV[64 * 576];   // qkv; reused as Y (out-proj result)
  __shared__ bhalf Pbuf[8 * 16 * 64];
  __shared__ float cA[192], cB[192];

  const int tid = threadIdx.x, lane = tid & 31, wave = tid >> 5;
  const int wid = blockIdx.x;
  const int b = wid >> 8, hwi = (wid >> 4) & 15, wwi = wid & 15;

  // fold groupnorm + time-affine into per-channel (a,b)
  for (int c = tid; c < 192; c += 256) {
    const int g = c / CPG;
    const float m = mean1[b * NGRP + g], r = rstd1[b * NGRP + g];
    const float sc = 1.f + p1[b * 384 + c];
    const float sf = p1[b * 384 + 192 + c];
    const float w = n1w[c] * r;
    cA[c] = w * sc;
    cB[c] = (n1b[c] - m * w) * sc + sf;
  }
  __syncthreads();

  // gather rolled window -> bf16 LDS (64 x 192)
  for (int t = tid; t < 64 * 192; t += 256) {
    const int c = t >> 6, l = t & 63;
    const int y = ((hwi * 8 + (l >> 3)) + SHF) & 127;
    const int x = ((wwi * 8 + (l & 7)) + SHF) & 127;
    const float v = xin[(((size_t)b * 192 + c) << 14) + (y << 7) + x];
    Xw[l * 192 + c] = (bhalf)(v * cA[c] + cB[c]);
  }
  __syncthreads();

  const v8f vzero = {};

  // QKV = Xw(64x192) @ Win^T(192x576): 4x36 tiles, 6 k-steps
  for (int t = wave; t < 144; t += 8) {
    const int mt = t & 3, nt = t >> 2;
    v8f acc = vzero;
#pragma unroll
    for (int k = 0; k < 6; ++k)
      acc = wmma_bf16(ldA(Xw, 192, mt * 16, k * 32, lane),
                      ldBt(w_in, 192, nt * 16, k * 32, lane), acc);
    const int n = nt * 16 + (lane & 15);
    const float bias = b_in[n];
    const float s = (n < 192) ? 0.17677669529663687f : 1.f;  // scale Q by 1/sqrt(32)
    const int m0 = mt * 16 + ((lane >> 4) << 3);
#pragma unroll
    for (int v = 0; v < 8; ++v) QKV[(m0 + v) * 576 + n] = (bhalf)((acc[v] + bias) * s);
  }
  __syncthreads();

  // attention: each wave owns 16-row strips of (head, mtile); 24 strips total
  for (int st = wave; st < 24; st += 8) {
    const int h = st >> 2, mt = st & 3;
    const v16bf aq = ldA(QKV, 576, mt * 16, h * 32, lane);
    v8f s0 = vzero, s1 = vzero, s2 = vzero, s3 = vzero;
    s0 = wmma_bf16(aq, ldBt(QKV, 576, 0,  192 + h * 32, lane), s0);
    s1 = wmma_bf16(aq, ldBt(QKV, 576, 16, 192 + h * 32, lane), s1);
    s2 = wmma_bf16(aq, ldBt(QKV, 576, 32, 192 + h * 32, lane), s2);
    s3 = wmma_bf16(aq, ldBt(QKV, 576, 48, 192 + h * 32, lane), s3);
    bhalf* P = &Pbuf[wave * 1024];
    const int n = lane & 15, hg = lane >> 4;
#pragma unroll
    for (int v = 0; v < 8; ++v) {
      float mx = fmaxf(fmaxf(s0[v], s1[v]), fmaxf(s2[v], s3[v]));
#pragma unroll
      for (int o = 1; o < 16; o <<= 1) mx = fmaxf(mx, __shfl_xor(mx, o, 32));
      const float e0 = __expf(s0[v] - mx), e1 = __expf(s1[v] - mx);
      const float e2 = __expf(s2[v] - mx), e3 = __expf(s3[v] - mx);
      float sm = e0 + e1 + e2 + e3;
#pragma unroll
      for (int o = 1; o < 16; o <<= 1) sm += __shfl_xor(sm, o, 32);
      const float inv = 1.f / sm;
      const int m = v + (hg << 3);
      P[m * 64 + n]      = (bhalf)(e0 * inv);
      P[m * 64 + 16 + n] = (bhalf)(e1 * inv);
      P[m * 64 + 32 + n] = (bhalf)(e2 * inv);
      P[m * 64 + 48 + n] = (bhalf)(e3 * inv);
    }
    // O_strip(16x32) = P(16x64) @ V_h(64x32)
    v8f o0 = vzero, o1 = vzero;
#pragma unroll
    for (int ks = 0; ks < 2; ++ks) {
      const v16bf ap = ldA(P, 64, 0, ks * 32, lane);
      o0 = wmma_bf16(ap, ldBn(QKV, 576, 384 + h * 32,      ks * 32, lane), o0);
      o1 = wmma_bf16(ap, ldBn(QKV, 576, 384 + h * 32 + 16, ks * 32, lane), o1);
    }
    const int m0 = mt * 16 + (hg << 3);
#pragma unroll
    for (int v = 0; v < 8; ++v) {
      Xw[(m0 + v) * 192 + h * 32 + n]      = (bhalf)o0[v];
      Xw[(m0 + v) * 192 + h * 32 + 16 + n] = (bhalf)o1[v];
    }
  }
  __syncthreads();

  // Y = O(64x192) @ Wout^T + b_out -> QKV region (bf16)
  for (int t = wave; t < 48; t += 8) {
    const int mt = t & 3, nt = t >> 2;
    v8f acc = vzero;
#pragma unroll
    for (int k = 0; k < 6; ++k)
      acc = wmma_bf16(ldA(Xw, 192, mt * 16, k * 32, lane),
                      ldBt(w_out, 192, nt * 16, k * 32, lane), acc);
    const int n = nt * 16 + (lane & 15);
    const float bias = b_out[n];
    const int m0 = mt * 16 + ((lane >> 4) << 3);
#pragma unroll
    for (int v = 0; v < 8; ++v) QKV[(m0 + v) * 192 + n] = (bhalf)(acc[v] + bias);
  }
  __syncthreads();

  // Z = Y @ Wproj^T + b_proj; roll back + residual, write x1
  for (int t = wave; t < 48; t += 8) {
    const int mt = t & 3, nt = t >> 2;
    v8f acc = vzero;
#pragma unroll
    for (int k = 0; k < 6; ++k)
      acc = wmma_bf16(ldA(QKV, 192, mt * 16, k * 32, lane),
                      ldBt(w_proj, 192, nt * 16, k * 32, lane), acc);
    const int n = nt * 16 + (lane & 15);
    const float bias = b_proj[n];
    const int m0 = mt * 16 + ((lane >> 4) << 3);
#pragma unroll
    for (int v = 0; v < 8; ++v) {
      const int l = m0 + v;
      const int y = ((hwi * 8 + (l >> 3)) + SHF) & 127;
      const int x = ((wwi * 8 + (l & 7)) + SHF) & 127;
      const size_t idx = (((size_t)b * 192 + n) << 14) + (y << 7) + x;
      x1[idx] = xin[idx] + acc[v] + bias;
    }
  }
}

// ---------------- conv3x3 (implicit GEMM over 9 taps) + SiLU + pooled --------
__global__ __launch_bounds__(256) void k_conv(
    const float* __restrict__ x1, const bhalf* __restrict__ w_cc,
    const float* __restrict__ b_cc, float* __restrict__ cc,
    float* __restrict__ pooled) {
  __shared__ bhalf tile[10 * 10 * 192];
  __shared__ float psum[192];
  const int tid = threadIdx.x, lane = tid & 31, wave = tid >> 5;
  const int id = blockIdx.x;
  const int b = id >> 8, ty = (id >> 4) & 15, tx = id & 15;
  const int y0 = ty * 8, x0 = tx * 8;
  for (int t = tid; t < 192; t += 256) psum[t] = 0.f;
  for (int t = tid; t < 10 * 10 * 192; t += 256) {
    const int c = t / 100, r = t % 100;
    const int ys = y0 + (r / 10) - 1, xs = x0 + (r % 10) - 1;
    float v = 0.f;
    if (ys >= 0 && ys < 128 && xs >= 0 && xs < 128)
      v = x1[(((size_t)b * 192 + c) << 14) + (ys << 7) + xs];
    tile[r * 192 + c] = (bhalf)v;
  }
  __syncthreads();
  const v8f vzero = {};
  v8f acc[6];
#pragma unroll
  for (int tt = 0; tt < 6; ++tt) acc[tt] = vzero;
  for (int tap = 0; tap < 9; ++tap) {
    const int ky = tap / 3, kx = tap % 3;
    const bhalf* wt = w_cc + tap * 192 * 192;
    for (int k = 0; k < 6; ++k) {
#pragma unroll
      for (int tt = 0; tt < 6; ++tt) {
        const int t = wave + 8 * tt;
        const int mt = t / 12, nt = t % 12;
        const int m = mt * 16 + (lane & 15);
        const int py = (m >> 3) + ky, px = (m & 7) + kx;
        const v16bf a =
            ldA_row(tile + (py * 10 + px) * 192 + k * 32 + ((lane >> 4) << 4));
        acc[tt] = wmma_bf16(a, ldBt(wt, 192, nt * 16, k * 32, lane), acc[tt]);
      }
    }
  }
#pragma unroll
  for (int tt = 0; tt < 6; ++tt) {
    const int t = wave + 8 * tt;
    const int mt = t / 12, nt = t % 12;
    const int n = nt * 16 + (lane & 15);
    const float bias = b_cc[n];
    const int m0 = mt * 16 + ((lane >> 4) << 3);
    float part = 0.f;
#pragma unroll
    for (int v = 0; v < 8; ++v) {
      const float z = acc[tt][v] + bias;
      const float s = z / (1.f + __expf(-z));
      part += s;
      const int p = m0 + v;
      cc[(((size_t)b * 192 + n) << 14) + ((y0 + (p >> 3)) << 7) + (x0 + (p & 7))] = s;
    }
    atomicAdd(&psum[n], part);
  }
  __syncthreads();
  for (int t = tid; t < 192; t += 256) atomicAdd(&pooled[b * 192 + t], psum[t]);
}

// ---------------- SE gate MLP ------------------------------------------------
__global__ void k_gate(const float* __restrict__ pooled, const float* __restrict__ g1_w,
                       const float* __restrict__ g1_b, const float* __restrict__ g2_w,
                       const float* __restrict__ g2_b, float* __restrict__ gate) {
  __shared__ float gh[Bsz][GATEH];
  const int tid = threadIdx.x;
  if (tid < Bsz * GATEH) {
    const int b = tid / GATEH, o = tid % GATEH;
    float acc = g1_b[o];
    for (int c = 0; c < 192; ++c)
      acc += (pooled[b * 192 + c] * (1.f / 16384.f)) * g1_w[o * 192 + c];
    gh[b][o] = acc / (1.f + __expf(-acc));
  }
  __syncthreads();
  for (int t = tid; t < Bsz * 192; t += blockDim.x) {
    const int b = t / 192, o = t % 192;
    float acc = g2_b[o];
    for (int c = 0; c < GATEH; ++c) acc += gh[b][c] * g2_w[o * GATEH + c];
    gate[t] = 1.f / (1.f + __expf(-acc));
  }
}

// ---------------- x2 = x1 + cc*gate (in place on x1) -------------------------
__global__ void k_se(float* __restrict__ x1, const float* __restrict__ cc,
                     const float* __restrict__ gate) {
  const size_t total = (size_t)Bsz * Cch * HWp;
  for (size_t i = (size_t)blockIdx.x * blockDim.x + threadIdx.x; i < total;
       i += (size_t)gridDim.x * blockDim.x) {
    const size_t bc = i >> 14;  // b*192 + c
    x1[i] = x1[i] + cc[i] * gate[bc];
  }
}

// ---------------- fused norm2 + affine + FFN + residual ----------------------
__global__ __launch_bounds__(256) void k_ffn(
    const float* __restrict__ x2, const float* __restrict__ p2,
    const float* __restrict__ n2w, const float* __restrict__ n2b,
    const float* __restrict__ mean2, const float* __restrict__ rstd2,
    const bhalf* __restrict__ w_f1, const float* __restrict__ b_f1,
    const bhalf* __restrict__ w_f2, const float* __restrict__ b_f2,
    float* __restrict__ out) {
  __shared__ bhalf Xt[64 * 192];
  __shared__ bhalf Hb[64 * 384];
  __shared__ float cA[192], cB[192];
  const int tid = threadIdx.x, lane = tid & 31, wave = tid >> 5;
  const int id = blockIdx.x;
  const int b = id >> 8;
  const int p0 = (id & 255) * 64;
  for (int c = tid; c < 192; c += 256) {
    const int g = c / CPG;
    const float m = mean2[b * NGRP + g], r = rstd2[b * NGRP + g];
    const float sc = 1.f + p2[b * 384 + c];
    const float sf = p2[b * 384 + 192 + c];
    const float w = n2w[c] * r;
    cA[c] = w * sc;
    cB[c] = (n2b[c] - m * w) * sc + sf;
  }
  __syncthreads();
  for (int t = tid; t < 64 * 192; t += 256) {
    const int c = t >> 6, l = t & 63;
    const float v = x2[(((size_t)b * 192 + c) << 14) + p0 + l];
    Xt[l * 192 + c] = (bhalf)(v * cA[c] + cB[c]);
  }
  __syncthreads();
  const v8f vzero = {};
  // H = silu(Xt @ f1^T + b): 4x24 tiles
  for (int t = wave; t < 96; t += 8) {
    const int mt = t & 3, nt = t >> 2;
    v8f acc = vzero;
#pragma unroll
    for (int k = 0; k < 6; ++k)
      acc = wmma_bf16(ldA(Xt, 192, mt * 16, k * 32, lane),
                      ldBt(w_f1, 192, nt * 16, k * 32, lane), acc);
    const int n = nt * 16 + (lane & 15);
    const float bias = b_f1[n];
    const int m0 = mt * 16 + ((lane >> 4) << 3);
#pragma unroll
    for (int v = 0; v < 8; ++v) {
      const float z = acc[v] + bias;
      Hb[(m0 + v) * 384 + n] = (bhalf)(z / (1.f + __expf(-z)));
    }
  }
  __syncthreads();
  // out = H @ f2^T + b + residual: 4x12 tiles, 12 k-steps
  for (int t = wave; t < 48; t += 8) {
    const int mt = t & 3, nt = t >> 2;
    v8f acc = vzero;
#pragma unroll
    for (int k = 0; k < 12; ++k)
      acc = wmma_bf16(ldA(Hb, 384, mt * 16, k * 32, lane),
                      ldBt(w_f2, 384, nt * 16, k * 32, lane), acc);
    const int n = nt * 16 + (lane & 15);
    const float bias = b_f2[n];
    const int m0 = mt * 16 + ((lane >> 4) << 3);
    const size_t base = (((size_t)b * 192 + n) << 14) + p0;
#pragma unroll
    for (int v = 0; v < 8; ++v) out[base + m0 + v] = x2[base + m0 + v] + acc[v] + bias;
  }
}

extern "C" void kernel_launch(void* const* d_in, const int* in_sizes, int n_in,
                              void* d_out, int out_size, void* d_ws, size_t ws_size,
                              hipStream_t stream) {
  (void)in_sizes; (void)n_in; (void)out_size; (void)ws_size;
  const float* x      = (const float*)d_in[0];
  const float* t_emb  = (const float*)d_in[1];
  const float* n1_w   = (const float*)d_in[2];
  const float* n1_b   = (const float*)d_in[3];
  const float* ta1_w  = (const float*)d_in[4];
  const float* ta1_b  = (const float*)d_in[5];
  const float* ain_w  = (const float*)d_in[6];
  const float* ain_b  = (const float*)d_in[7];
  const float* aout_w = (const float*)d_in[8];
  const float* aout_b = (const float*)d_in[9];
  const float* proj_w = (const float*)d_in[10];
  const float* proj_b = (const float*)d_in[11];
  const float* cc_w   = (const float*)d_in[12];
  const float* cc_b   = (const float*)d_in[13];
  const float* g1_w   = (const float*)d_in[14];
  const float* g1_b   = (const float*)d_in[15];
  const float* g2_w   = (const float*)d_in[16];
  const float* g2_b   = (const float*)d_in[17];
  const float* n2_w   = (const float*)d_in[18];
  const float* n2_b   = (const float*)d_in[19];
  const float* ta2_w  = (const float*)d_in[20];
  const float* ta2_b  = (const float*)d_in[21];
  const float* f1_w   = (const float*)d_in[22];
  const float* f1_b   = (const float*)d_in[23];
  const float* f2_w   = (const float*)d_in[24];
  const float* f2_b   = (const float*)d_in[25];

  const size_t NBIG = (size_t)Bsz * Cch * HWp;  // 25,165,824
  float* x1 = (float*)d_ws;
  float* cc = x1 + NBIG;
  bhalf* w_in   = (bhalf*)(cc + NBIG);
  bhalf* w_out  = w_in + 576 * 192;
  bhalf* w_proj = w_out + 192 * 192;
  bhalf* w_f1   = w_proj + 192 * 192;
  bhalf* w_f2   = w_f1 + 384 * 192;
  bhalf* w_cc   = w_f2 + 192 * 384;
  float* p1     = (float*)(w_cc + 9 * 192 * 192);
  float* p2     = p1 + Bsz * 384;
  float* mean1  = p2 + Bsz * 384;
  float* rstd1  = mean1 + 256;
  float* mean2  = rstd1 + 256;
  float* rstd2  = mean2 + 256;
  float* pooled = rstd2 + 256;
  float* gate   = pooled + Bsz * 192;

  k_prep<<<256, 256, 0, stream>>>(ain_w, aout_w, proj_w, f1_w, f2_w, cc_w, w_in,
                                  w_out, w_proj, w_f1, w_f2, w_cc, pooled);
  k_taffine<<<24, 256, 0, stream>>>(t_emb, ta1_w, ta1_b, ta2_w, ta2_b, p1, p2);
  k_gnstats<<<256, 256, 0, stream>>>(x, mean1, rstd1);
  k_attn<<<2048, 256, 0, stream>>>(x, p1, n1_w, n1_b, mean1, rstd1, w_in, ain_b,
                                   w_out, aout_b, w_proj, proj_b, x1);
  k_conv<<<2048, 256, 0, stream>>>(x1, w_cc, cc_b, cc, pooled);
  k_gate<<<1, 256, 0, stream>>>(pooled, g1_w, g1_b, g2_w, g2_b, gate);
  k_se<<<4096, 256, 0, stream>>>(x1, cc, gate);
  k_gnstats<<<256, 256, 0, stream>>>(x1, mean2, rstd2);
  k_ffn<<<2048, 256, 0, stream>>>(x1, p2, n2_w, n2_b, mean2, rstd2, w_f1, f1_b,
                                  w_f2, f2_b, (float*)d_out);
}
